// MultiHeadLatentAttention_21878563406426
// MI455X (gfx1250) — matrix operand order repaired and checked by reading, not compiled, not run on hardware
//
#include <hip/hip_runtime.h>

#define BN   2
#define TN   2048
#define DN   1024
#define HN   16
#define HDN  64
#define LATN 256
#define BTN  (BN * TN)   // 4096 rows
#define AW   4           // waves per attention block

typedef __attribute__((ext_vector_type(16))) __bf16 v16bf;
typedef __attribute__((ext_vector_type(8)))  __bf16 v8bf;
typedef __attribute__((ext_vector_type(8)))  float  v8f;

// ---------------- helpers ----------------

__device__ __forceinline__ unsigned short f2bf(float x) {
  unsigned int u = __float_as_uint(x);
  unsigned int r = u + 0x7FFFu + ((u >> 16) & 1u);   // round-to-nearest-even
  return (unsigned short)(r >> 16);
}

__device__ __forceinline__ v8f wmma_bf16(v16bf a, v16bf b, v8f c) {
  // D(f32 16x16) = A(bf16 16x32) * B(bf16 32x16) + C
  return __builtin_amdgcn_wmma_f32_16x16x32_bf16(false, a, false, b, (short)0, c,
                                                 false, false);
}

// A-operand fragment (16x32 bf16) from row-major A[row][k], ld = row stride.
// lanes 0-15: row=lane, elems 0-7 = K[k0..k0+7], elems 8-15 = K[k0+16..k0+23]
// lanes16-31: row=lane-16, K[k0+8..k0+15] and K[k0+24..k0+31]
__device__ __forceinline__ v16bf load_a_frag(const unsigned short* base, int ld,
                                             int k0, int lane) {
  int r   = lane & 15;
  int sel = (lane >> 4) & 1;
  const unsigned short* p = base + (size_t)r * ld + k0 + sel * 8;
  v8bf c0 = *reinterpret_cast<const v8bf*>(p);
  v8bf c1 = *reinterpret_cast<const v8bf*>(p + 16);
  return __builtin_shufflevector(c0, c1, 0, 1, 2, 3, 4, 5, 6, 7,
                                 8, 9, 10, 11, 12, 13, 14, 15);
}

// B-operand fragment (32x16 bf16) from Bt[n][k] row-major.
// lanes 0-15: col n=lane, elems 0-7 = K[k0..k0+7], elems 8-15 = K[k0+16..k0+23]
// lanes16-31: col n=lane-16, K[k0+8..k0+15] and K[k0+24..k0+31]
__device__ __forceinline__ v16bf load_b_frag(const unsigned short* base, int ld,
                                             int k0, int lane) {
  int r   = lane & 15;
  int sel = (lane >> 4) & 1;
  const unsigned short* p = base + (size_t)r * ld + k0 + sel * 8;
  v8bf c0 = *reinterpret_cast<const v8bf*>(p);
  v8bf c1 = *reinterpret_cast<const v8bf*>(p + 16);
  return __builtin_shufflevector(c0, c1, 0, 1, 2, 3, 4, 5, 6, 7,
                                 8, 9, 10, 11, 12, 13, 14, 15);
}

// ---------------- conversion kernels ----------------

__global__ void __launch_bounds__(256) cvt4_f32_bf16(const float* __restrict__ in,
                                                     unsigned short* __restrict__ out,
                                                     int n4) {
  int i = blockIdx.x * blockDim.x + threadIdx.x;
  if (i >= n4) return;
  float4 v = reinterpret_cast<const float4*>(in)[i];
  ushort4 o;
  o.x = f2bf(v.x); o.y = f2bf(v.y); o.z = f2bf(v.z); o.w = f2bf(v.w);
  reinterpret_cast<ushort4*>(out)[i] = o;
}

// W[k][n] f32  ->  Wt[n][k] bf16
__global__ void __launch_bounds__(256) cvtT_f32_bf16(const float* __restrict__ in,
                                                     unsigned short* __restrict__ out,
                                                     int K, int N) {
  int idx = blockIdx.x * blockDim.x + threadIdx.x;
  if (idx >= K * N) return;
  int k = idx / N;
  int n = idx - k * N;
  out[(size_t)n * K + k] = f2bf(in[idx]);
}

// ---------------- WMMA GEMM ----------------
// C[M,N] = A[M,K]*B[K,N]; A row-major bf16, Bt[n][k] bf16.
// One wave computes a 32x64 strip (2 A frags x 4 B frags = 8 WMMA / K-step).
// MODE 0: f32 out; MODE 1: bf16 out; MODE 2: bf16 scattered to vT[(b*H+h)*HD+hd][t]
template <int MODE>
__global__ void __launch_bounds__(256) gemm_k(const unsigned short* __restrict__ A,
                                              const unsigned short* __restrict__ Bt,
                                              void* __restrict__ Cout,
                                              int M, int N, int K) {
  int lane = threadIdx.x & 31;
  int wave = blockIdx.x * (blockDim.x >> 5) + (threadIdx.x >> 5);
  int nb64 = N >> 6;
  int mt = wave / nb64;
  int nb = wave - mt * nb64;
  if (mt * 32 >= M) return;

  const unsigned short* Abase0 = A + (size_t)(mt * 32) * K;
  const unsigned short* Abase1 = Abase0 + (size_t)16 * K;
  const unsigned short* Bbase  = Bt + (size_t)(nb * 64) * K;

  v8f acc[2][4] = {};
  for (int k0 = 0; k0 < K; k0 += 32) {
    if (k0 + 32 < K) {
      __builtin_prefetch(Abase0 + (size_t)(lane & 15) * K + k0 + 32, 0, 3);
      __builtin_prefetch(Bbase + (size_t)(lane & 15) * K + k0 + 32, 0, 3);
    }
    v16bf a0 = load_a_frag(Abase0, K, k0, lane);
    v16bf a1 = load_a_frag(Abase1, K, k0, lane);
#pragma unroll
    for (int j = 0; j < 4; ++j) {
      v16bf b = load_b_frag(Bbase + (size_t)(j * 16) * K, K, k0, lane);
      acc[0][j] = wmma_bf16(a0, b, acc[0][j]);
      acc[1][j] = wmma_bf16(a1, b, acc[1][j]);
    }
  }

  int half = lane >> 4;
  int col0 = lane & 15;
#pragma unroll
  for (int s = 0; s < 2; ++s) {
#pragma unroll
    for (int r = 0; r < 8; ++r) {
      int row = mt * 32 + s * 16 + r + half * 8;
#pragma unroll
      for (int j = 0; j < 4; ++j) {
        int col = nb * 64 + j * 16 + col0;
        float v = acc[s][j][r];
        if (MODE == 0) {
          reinterpret_cast<float*>(Cout)[(size_t)row * N + col] = v;
        } else if (MODE == 1) {
          reinterpret_cast<unsigned short*>(Cout)[(size_t)row * N + col] = f2bf(v);
        } else {
          int bb = row >> 11;
          int t  = row & (TN - 1);
          int hh = col >> 6;
          int hd = col & (HDN - 1);
          reinterpret_cast<unsigned short*>(Cout)
              [(((size_t)(bb * HN + hh)) * HDN + hd) * TN + t] = f2bf(v);
        }
      }
    }
  }
}

// ---------------- LayerNorm over LAT=256 ----------------

__global__ void __launch_bounds__(256) ln_k(const float* __restrict__ in,
                                            const float* __restrict__ gamma,
                                            const float* __restrict__ beta,
                                            unsigned short* __restrict__ out) {
  int row  = blockIdx.x;
  int tid  = threadIdx.x;
  int lane = tid & 31;
  int w    = tid >> 5;
  __shared__ float red[8];

  float x = in[(size_t)row * LATN + tid];
  float s = x;
#pragma unroll
  for (int m = 16; m >= 1; m >>= 1) s += __shfl_xor(s, m, 32);
  if (lane == 0) red[w] = s;
  __syncthreads();
  float tot = 0.f;
#pragma unroll
  for (int i = 0; i < 8; ++i) tot += red[i];
  float mu = tot * (1.0f / LATN);
  float d  = x - mu;
  float s2 = d * d;
#pragma unroll
  for (int m = 16; m >= 1; m >>= 1) s2 += __shfl_xor(s2, m, 32);
  __syncthreads();
  if (lane == 0) red[w] = s2;
  __syncthreads();
  float var = 0.f;
#pragma unroll
  for (int i = 0; i < 8; ++i) var += red[i];
  var *= (1.0f / LATN);
  float y = d * rsqrtf(var + 1e-5f) * gamma[tid] + beta[tid];
  out[(size_t)row * LATN + tid] = f2bf(y);
}

// ---------------- causal flash attention ----------------
// One wave per (b, h, 16-query tile). fp32 online softmax.
// K tiles are double-buffered in LDS via async global->LDS copies (ASYNCcnt).

// Copy a 32-key x 64-hd bf16 tile (4KB) into LDS: 8 async b128 ops, 512B each.
__device__ __forceinline__ void async_copy_k_tile(const unsigned short* gsrc,
                                                  unsigned ldsbase, int lane) {
  int chunk = lane & 7;    // 8 x 16B chunks per 128B key-row
  int krow  = lane >> 3;   // 4 key-rows per instruction
#pragma unroll
  for (int i = 0; i < 8; ++i) {
    int key = krow + i * 4;
    unsigned long long g =
        (unsigned long long)(size_t)(gsrc + (size_t)key * DN + chunk * 8);
    unsigned l = ldsbase + (unsigned)(key * 128 + chunk * 16);
    asm volatile("global_load_async_to_lds_b128 %0, %1, off"
                 :: "v"(l), "v"(g) : "memory");
  }
}

__global__ void __launch_bounds__(128) attn_k(const unsigned short* __restrict__ qb,
                                              const unsigned short* __restrict__ kb,
                                              const unsigned short* __restrict__ vT,
                                              unsigned short* __restrict__ ob) {
  __shared__ __align__(16) unsigned short kst[AW][2][32 * HDN]; // 2 x 4KB per wave
  __shared__ __align__(16) unsigned short lp[AW][16 * 32];      // P tile per wave
  int wave = threadIdx.x >> 5;
  int lane = threadIdx.x & 31;
  int job  = blockIdx.x * AW + wave;       // 0 .. B*H*(T/16)-1
  int qt   = job & ((TN / 16) - 1);
  int bh   = job >> 7;
  int h    = bh & (HN - 1);
  int b    = bh >> 4;
  int half = lane >> 4;
  int col0 = lane & 15;

  const unsigned short* qbase = qb + ((size_t)(b * TN + qt * 16)) * DN + h * HDN;
  v16bf aq0 = load_a_frag(qbase, DN, 0, lane);
  v16bf aq1 = load_a_frag(qbase, DN, 32, lane);

  float mrow[8], lrow[8];
  v8f acc[4] = {};
#pragma unroll
  for (int r = 0; r < 8; ++r) { mrow[r] = -1e30f; lrow[r] = 0.f; }

  const unsigned short* kbh = kb + ((size_t)b * TN) * DN + h * HDN;
  const unsigned short* vbh = vT + ((size_t)(b * HN + h)) * HDN * TN;
  unsigned short* myp = &lp[wave][0];
  // low 32 bits of a generic LDS pointer == LDS byte offset (aperture in [63:32])
  unsigned kls[2];
  kls[0] = (unsigned)(size_t)&kst[wave][0][0];
  kls[1] = (unsigned)(size_t)&kst[wave][1][0];

  int nkeys = qt * 16 + 16;
  async_copy_k_tile(kbh, kls[0], lane);             // prefetch first tile

  for (int kb0 = 0; kb0 < nkeys; kb0 += 32) {
    int buf = (kb0 >> 5) & 1;
    if (kb0 + 32 < nkeys) {
      async_copy_k_tile(kbh + (size_t)(kb0 + 32) * DN, kls[buf ^ 1], lane);
      asm volatile("s_wait_asynccnt 0x8" ::: "memory");  // current tile resident
    } else {
      asm volatile("s_wait_asynccnt 0x0" ::: "memory");
    }
    // ---- S = Q K^T for 32 keys from LDS (two 16x16 C tiles) ----
    const unsigned short* kt0 = &kst[wave][buf][0];
    const unsigned short* kt1 = kt0 + 16 * HDN;
    v8f s0 = {}, s1 = {};
    s0 = wmma_bf16(aq0, load_b_frag(kt0, HDN, 0, lane), s0);
    s0 = wmma_bf16(aq1, load_b_frag(kt0, HDN, 32, lane), s0);
    s1 = wmma_bf16(aq0, load_b_frag(kt1, HDN, 0, lane), s1);
    s1 = wmma_bf16(aq1, load_b_frag(kt1, HDN, 32, lane), s1);

    // ---- mask + online softmax (fp32) ----
#pragma unroll
    for (int r = 0; r < 8; ++r) {
      int qrow = qt * 16 + r + half * 8;
      float a0 = s0[r] * 0.125f;                // 1/sqrt(64)
      float a1 = s1[r] * 0.125f;
      if (kb0 + col0 > qrow)      a0 = -1e30f;
      if (kb0 + 16 + col0 > qrow) a1 = -1e30f;
      float rm = fmaxf(a0, a1);
      rm = fmaxf(rm, __shfl_xor(rm, 1, 32));
      rm = fmaxf(rm, __shfl_xor(rm, 2, 32));
      rm = fmaxf(rm, __shfl_xor(rm, 4, 32));
      rm = fmaxf(rm, __shfl_xor(rm, 8, 32));
      float mn    = fmaxf(mrow[r], rm);
      float alpha = __expf(mrow[r] - mn);
      mrow[r] = mn;
      float p0 = __expf(a0 - mn);
      float p1 = __expf(a1 - mn);
      float rs = p0 + p1;
      rs += __shfl_xor(rs, 1, 32);
      rs += __shfl_xor(rs, 2, 32);
      rs += __shfl_xor(rs, 4, 32);
      rs += __shfl_xor(rs, 8, 32);
      lrow[r] = lrow[r] * alpha + rs;
      acc[0][r] *= alpha;
      acc[1][r] *= alpha;
      acc[2][r] *= alpha;
      acc[3][r] *= alpha;
      int lr = r + half * 8;
      myp[lr * 32 + col0]      = f2bf(p0);   // C layout -> LDS [row][key]
      myp[lr * 32 + 16 + col0] = f2bf(p1);
    }
    // ---- reload P as A-operand (wave-local LDS ops are in-order) ----
    v16bf pa;
    {
      int pr  = lane & 15;
      const unsigned short* pp = myp + pr * 32 + half * 8;
      v8bf c0 = *reinterpret_cast<const v8bf*>(pp);
      v8bf c1 = *reinterpret_cast<const v8bf*>(pp + 16);
      pa = __builtin_shufflevector(c0, c1, 0, 1, 2, 3, 4, 5, 6, 7,
                                   8, 9, 10, 11, 12, 13, 14, 15);
    }
    // ---- O += P * V  (V pre-transposed -> contiguous B fragments) ----
#pragma unroll
    for (int j = 0; j < 4; ++j) {
      v16bf bv = load_b_frag(vbh + (size_t)(j * 16) * TN, TN, kb0, lane);
      acc[j] = wmma_bf16(pa, bv, acc[j]);
    }
  }

  // ---- epilogue: normalize and store bf16 ----
  unsigned short* obase = ob + ((size_t)(b * TN + qt * 16)) * DN + h * HDN;
#pragma unroll
  for (int r = 0; r < 8; ++r) {
    float inv = 1.0f / lrow[r];
    int row = r + half * 8;
#pragma unroll
    for (int j = 0; j < 4; ++j) {
      obase[(size_t)row * DN + j * 16 + col0] = f2bf(acc[j][r] * inv);
    }
  }
}

// ---------------- launch ----------------

extern "C" void kernel_launch(void* const* d_in, const int* in_sizes, int n_in,
                              void* d_out, int out_size, void* d_ws, size_t ws_size,
                              hipStream_t stream) {
  (void)in_sizes; (void)n_in; (void)out_size; (void)ws_size;
  const float* x     = (const float*)d_in[0];
  const float* Wq    = (const float*)d_in[1];
  const float* Wdown = (const float*)d_in[2];
  const float* Wkup  = (const float*)d_in[3];
  const float* Wvup  = (const float*)d_in[4];
  const float* gam   = (const float*)d_in[5];
  const float* bet   = (const float*)d_in[6];
  const float* Wo    = (const float*)d_in[7];
  float* out = (float*)d_out;

  char* ws = (char*)d_ws;
  size_t off = 0;
  auto alloc = [&](size_t bytes) -> void* {
    void* p = ws + off;
    off += (bytes + 255) & ~(size_t)255;
    return p;
  };
  unsigned short* xb     = (unsigned short*)alloc((size_t)BTN * DN * 2);
  unsigned short* wqT    = (unsigned short*)alloc((size_t)DN * DN * 2);
  unsigned short* wdT    = (unsigned short*)alloc((size_t)LATN * DN * 2);
  unsigned short* wkT    = (unsigned short*)alloc((size_t)DN * LATN * 2);
  unsigned short* wvT    = (unsigned short*)alloc((size_t)DN * LATN * 2);
  unsigned short* woT    = (unsigned short*)alloc((size_t)DN * DN * 2);
  unsigned short* qbuf   = (unsigned short*)alloc((size_t)BTN * DN * 2);
  float*          latraw = (float*)alloc((size_t)BTN * LATN * 4);
  unsigned short* latb   = (unsigned short*)alloc((size_t)BTN * LATN * 2);
  unsigned short* kbuf   = (unsigned short*)alloc((size_t)BTN * DN * 2);
  unsigned short* vTb    = (unsigned short*)alloc((size_t)BN * HN * HDN * TN * 2);
  unsigned short* aob    = xb;  // reuse: x dead after q/latent GEMMs

  cvt4_f32_bf16<<<(BTN * DN / 4 + 255) / 256, 256, 0, stream>>>(x, xb, BTN * DN / 4);
  cvtT_f32_bf16<<<(DN * DN + 255) / 256, 256, 0, stream>>>(Wq, wqT, DN, DN);
  cvtT_f32_bf16<<<(DN * LATN + 255) / 256, 256, 0, stream>>>(Wdown, wdT, DN, LATN);
  cvtT_f32_bf16<<<(LATN * DN + 255) / 256, 256, 0, stream>>>(Wkup, wkT, LATN, DN);
  cvtT_f32_bf16<<<(LATN * DN + 255) / 256, 256, 0, stream>>>(Wvup, wvT, LATN, DN);
  cvtT_f32_bf16<<<(DN * DN + 255) / 256, 256, 0, stream>>>(Wo, woT, DN, DN);

  // q = x @ Wq (bf16)
  gemm_k<1><<<(BTN / 32) * (DN / 64) / 8, 256, 0, stream>>>(xb, wqT, qbuf, BTN, DN, DN);
  // lat_raw = x @ Wdown (f32)
  gemm_k<0><<<(BTN / 32) * (LATN / 64) / 8, 256, 0, stream>>>(xb, wdT, latraw, BTN, LATN, DN);
  // latent = LN(lat_raw) (bf16)
  ln_k<<<BTN, 256, 0, stream>>>(latraw, gam, bet, latb);
  // k = latent @ Wkup (bf16 row-major)
  gemm_k<1><<<(BTN / 32) * (DN / 64) / 8, 256, 0, stream>>>(latb, wkT, kbuf, BTN, DN, LATN);
  // v = latent @ Wvup (bf16 -> vT per head)
  gemm_k<2><<<(BTN / 32) * (DN / 64) / 8, 256, 0, stream>>>(latb, wvT, vTb, BTN, DN, LATN);
  // attention (4 waves / 128-thread block)
  attn_k<<<(BN * HN * (TN / 16)) / AW, 128, 0, stream>>>(qbuf, kbuf, vTb, aob);
  // out = attn_out @ Wo (f32)
  gemm_k<0><<<(BTN / 32) * (DN / 64) / 8, 256, 0, stream>>>(aob, woT, out, BTN, DN, DN);
}
